// BPTRUCell_25116968747067
// MI455X (gfx1250) — matrix operand order repaired and checked by reading, not compiled, not run on hardware
//
#include <hip/hip_runtime.h>
#include <hip/hip_bf16.h>
#include <math.h>

#define LSEQ 512
#define NBATCH 64
#define ISZ 512
#define HSZ 1024
#define SCAN_BLOCKS 32          // H / 32 columns each
#define NCOL 32                 // columns per scan block
#define LDSW (HSZ + 8)          // padded LDS row stride (elements) to break bank conflicts

typedef __attribute__((ext_vector_type(16))) __bf16 v16bf;
typedef __attribute__((ext_vector_type(8)))  __bf16 v8bf;
typedef __attribute__((ext_vector_type(8)))  float  v8f;

__device__ __forceinline__ float sigmoidf_(float x) {
    return 1.0f / (1.0f + __expf(-x));
}

// A-fragment (16x32 bf16, row-major bf16 source).
// lanes 0-15: K = {0..15}; lanes 16-31: K = {16..31}; row = lane&15.
__device__ __forceinline__ v16bf load_a_bf16(const __bf16* __restrict__ rowp, int k0, int half) {
    const __bf16* q = rowp + k0 + half * 16;
    v8bf lo = *(const v8bf*)q;
    v8bf hi = *(const v8bf*)(q + 8);
    return __builtin_shufflevector(lo, hi, 0, 1, 2, 3, 4, 5, 6, 7,
                                           8, 9, 10, 11, 12, 13, 14, 15);
}

// B-fragment (32x16 bf16) from global: B[k][n] = W[n][k]; contiguous 16-element run.
__device__ __forceinline__ v16bf load_b_bf16(const __bf16* __restrict__ wrow, int k0, int half) {
    return *(const v16bf*)(wrow + k0 + half * 16);
}

// B-fragment from LDS (two 16B ds loads; base = this lane's column row in LDS).
__device__ __forceinline__ v16bf load_b_lds(const __bf16* base, int k0, int half) {
    const __bf16* q = base + k0 + half * 16;
    v8bf lo = *(const v8bf*)q;
    v8bf hi = *(const v8bf*)(q + 8);
    return __builtin_shufflevector(lo, hi, 0, 1, 2, 3, 4, 5, 6, 7,
                                           8, 9, 10, 11, 12, 13, 14, 15);
}

// ---------------------------------------------------------------------------
// One-time f32 -> bf16 conversion (n multiple of 4).
// ---------------------------------------------------------------------------
__global__ void bptru_f32_to_bf16(const float* __restrict__ src,
                                  __bf16* __restrict__ dst, int n) {
    int i = (blockIdx.x * blockDim.x + threadIdx.x) * 4;
    if (i + 3 < n) {
        float4 v = *(const float4*)(src + i);
        dst[i + 0] = (__bf16)v.x;
        dst[i + 1] = (__bf16)v.y;
        dst[i + 2] = (__bf16)v.z;
        dst[i + 3] = (__bf16)v.w;
    }
}

// ---------------------------------------------------------------------------
// Phase 1: xh_all[l, b, g] = sum_i x[b, l, i] * W_u[g, i]   (bf16 WMMA, f32 acc)
// Grid: (N/128, M/64). Block = 128 threads (4 waves). Each wave: 16 rows x 128 cols.
// ---------------------------------------------------------------------------
__global__ __launch_bounds__(128) void bptru_gemm_xu(const __bf16* __restrict__ xb,
                                                     const __bf16* __restrict__ Wub,
                                                     float* __restrict__ xh) {
    const int lane   = threadIdx.x & 31;
    const int wave   = threadIdx.x >> 5;
    const int n_base = blockIdx.x * 128;
    const int m_base = blockIdx.y * 64 + wave * 16;   // flat row m = l*64 + b
    const int l  = m_base >> 6;
    const int b0 = m_base & 63;
    const int half = lane >> 4;

    const __bf16* arow = xb + ((size_t)(b0 + (lane & 15)) * LSEQ + l) * ISZ;
    const __bf16* wrow[8];
#pragma unroll
    for (int t = 0; t < 8; ++t)
        wrow[t] = Wub + (size_t)(n_base + t * 16 + (lane & 15)) * ISZ;

    v8f acc[8] = {};

    for (int k0 = 0; k0 < ISZ; k0 += 32) {
        v16bf a = load_a_bf16(arow, k0, half);
#pragma unroll
        for (int t = 0; t < 8; ++t) {
            acc[t] = __builtin_amdgcn_wmma_f32_16x16x32_bf16(
                false, a, false, load_b_bf16(wrow[t], k0, half),
                (short)0, acc[t], false, false);
        }
    }

    const int rbase = m_base + half * 8;
#pragma unroll
    for (int t = 0; t < 8; ++t) {
        const int col = n_base + t * 16 + (lane & 15);
#pragma unroll
        for (int v = 0; v < 8; ++v) {
            xh[(size_t)(rbase + v) * (4 * HSZ) + col] = acc[t][v];
        }
    }
}

// ---------------------------------------------------------------------------
// Phase 2: persistent scan kernel.  One launch for all 512 timesteps.
// 32 blocks x 128 threads.  Block owns a 32-column strip; W_c/W_h columns live
// in LDS (bf16, ~129 KB of CDNA5's 320KB-per-WGP LDS) for all 512 steps.
// f32 state lives in REGISTERS across timesteps (each wave's D-tile element is
// the same (row,col) it updated last step); only bf16 state goes through
// memory for the next step's WMMA A-fragments.  Grid-wide release/acquire
// barrier between steps; xh gate inputs prefetched before the GEMM to overlap
// HBM latency with the WMMA loop.
// ---------------------------------------------------------------------------
__global__ __launch_bounds__(128) void bptru_scan(
    const float*  __restrict__ xh,     // [L*B, 4H]
    __bf16* __restrict__ cb0, __bf16* __restrict__ hb0,
    __bf16* __restrict__ cb1, __bf16* __restrict__ hb1,
    const __bf16* __restrict__ Wcb,    // [H, H] bf16
    const __bf16* __restrict__ Whb,
    const float* __restrict__ b_f, const float* __restrict__ v_f,
    const float* __restrict__ b_r, const float* __restrict__ v_r,
    const float* __restrict__ b_o,
    float* __restrict__ out,           // d_out: [B, L, H]
    float* __restrict__ tail,          // d_out + B*L*H : hT then cT, [B,H] each
    unsigned* __restrict__ counter) {
    extern __shared__ char smem[];
    __bf16* ldsc = (__bf16*)smem;            // [NCOL][LDSW]
    __bf16* ldsh = ldsc + NCOL * LDSW;

    const int lane = threadIdx.x & 31;
    const int wave = threadIdx.x >> 5;
    const int n0   = blockIdx.x * NCOL;
    const int m0   = wave * 16;
    const int half = lane >> 4;

    // ---- stage this block's weight columns into LDS (once) ----
    for (int idx = threadIdx.x; idx < NCOL * (HSZ / 8); idx += blockDim.x) {
        const int n  = idx >> 7;           // / (HSZ/8)
        const int kc = (idx & 127) * 8;
        *(v8bf*)&ldsc[n * LDSW + kc] = *(const v8bf*)&Wcb[(size_t)(n0 + n) * HSZ + kc];
        *(v8bf*)&ldsh[n * LDSW + kc] = *(const v8bf*)&Whb[(size_t)(n0 + n) * HSZ + kc];
    }
    __syncthreads();

    // per-lane LDS B rows for the two 16-col tiles
    const __bf16* bcrow[2] = { ldsc + (0 * 16 + (lane & 15)) * LDSW,
                               ldsc + (1 * 16 + (lane & 15)) * LDSW };
    const __bf16* bhrow[2] = { ldsh + (0 * 16 + (lane & 15)) * LDSW,
                               ldsh + (1 * 16 + (lane & 15)) * LDSW };

    // loop-invariant gate parameters for this lane's two columns
    const int colt[2] = { n0 + (lane & 15), n0 + 16 + (lane & 15) };
    float bo_[2], bff_[2], vff_[2], brr_[2], vrr_[2];
#pragma unroll
    for (int t = 0; t < 2; ++t) {
        bo_[t]  = b_o[colt[t]];
        bff_[t] = b_f[colt[t]];  vff_[t] = v_f[colt[t]];
        brr_[t] = b_r[colt[t]];  vrr_[t] = v_r[colt[t]];
    }

    // f32 state in registers (this wave's own D-tile elements, all timesteps)
    float creg[2][8] = {};
    float hreg[2][8] = {};

    const int rbase = m0 + half * 8;

    for (int l = 0; l < LSEQ; ++l) {
        const int cur = l & 1;
        const __bf16* cb_in = cur ? cb1 : cb0;
        const __bf16* hb_in = cur ? hb1 : hb0;
        __bf16* cb_out = cur ? cb0 : cb1;
        __bf16* hb_out = cur ? hb0 : hb1;

        // prefetch this step's xh gate lines (16 rows x 4 slices; a block's 32
        // cols are one 128B line).  Lane covers (row = m0+(lane&15), slice):
        {
            const int prow = m0 + (lane & 15);
            const float* pb = xh + ((size_t)l * NBATCH + prow) * (4 * HSZ) + n0;
            __builtin_prefetch(pb + (size_t)half * HSZ, 0, 1);
            __builtin_prefetch(pb + (size_t)(2 + half) * HSZ, 0, 1);
        }

        const __bf16* crow = cb_in + (size_t)(m0 + (lane & 15)) * HSZ;
        const __bf16* hrow = hb_in + (size_t)(m0 + (lane & 15)) * HSZ;

        v8f acc[2] = {};
        for (int k0 = 0; k0 < HSZ; k0 += 32) {
            v16bf ac = load_a_bf16(crow, k0, half);
            v16bf ah = load_a_bf16(hrow, k0, half);
#pragma unroll
            for (int t = 0; t < 2; ++t) {
                acc[t] = __builtin_amdgcn_wmma_f32_16x16x32_bf16(
                    false, ac, false, load_b_lds(bcrow[t], k0, half),
                    (short)0, acc[t], false, false);
                acc[t] = __builtin_amdgcn_wmma_f32_16x16x32_bf16(
                    false, ah, false, load_b_lds(bhrow[t], k0, half),
                    (short)0, acc[t], false, false);
            }
        }

        // ---- gates / state update for this wave's 16 rows x 32 cols ----
#pragma unroll
        for (int t = 0; t < 2; ++t) {
            const int col = colt[t];
#pragma unroll
            for (int v = 0; v < 8; ++v) {
                const int row = rbase + v;                 // batch index
                const size_t idx = (size_t)row * HSZ + col;
                const float c = creg[t][v];
                const float h = hreg[t][v];
                const float* xrow = xh + ((size_t)l * NBATCH + row) * (4 * HSZ);
                const float x1 = xrow[col];
                const float x2 = xrow[HSZ + col];
                const float x3 = xrow[2 * HSZ + col];
                const float x4 = xrow[3 * HSZ + col];
                const float o  = sigmoidf_(acc[t][v] + bo_[t]);
                const float f  = sigmoidf_(x1 + vff_[t] * c + bff_[t]);
                const float r  = sigmoidf_(x2 + vrr_[t] * c + brr_[t]);
                const float nc = tanhf(f * c + (1.0f - f) * x3 + o * h);
                const float nh = tanhf(r * h + (1.0f - r) * x4 + o * c);
                creg[t][v] = nc;
                hreg[t][v] = nh;
                cb_out[idx] = (__bf16)nc;
                hb_out[idx] = (__bf16)nh;
                out[((size_t)row * LSEQ + l) * HSZ + col] = nh;
                if (l == LSEQ - 1) {
                    tail[idx]                        = nh;   // hT
                    tail[(size_t)NBATCH * HSZ + idx] = nc;   // cT
                }
            }
        }

        // ---- grid-wide barrier: release bf16 state writes, wait for all blocks ----
        __threadfence();
        __syncthreads();
        if (threadIdx.x == 0) {
            __hip_atomic_fetch_add(counter, 1u, __ATOMIC_RELEASE, __HIP_MEMORY_SCOPE_AGENT);
            const unsigned target = (unsigned)(l + 1) * gridDim.x;
            while (__hip_atomic_load(counter, __ATOMIC_ACQUIRE, __HIP_MEMORY_SCOPE_AGENT) < target) {
                __builtin_amdgcn_s_sleep(2);
            }
        }
        __syncthreads();
    }
}

__global__ void bptru_init(__bf16* __restrict__ cb0, __bf16* __restrict__ hb0,
                           unsigned* __restrict__ counter, int n) {
    int i = blockIdx.x * blockDim.x + threadIdx.x;
    if (i == 0) *counter = 0u;           // re-zero every call (graph replay safe)
    if (i < n) {
        cb0[i] = (__bf16)0.0f;
        hb0[i] = (__bf16)0.0f;
    }
}

extern "C" void kernel_launch(void* const* d_in, const int* in_sizes, int n_in,
                              void* d_out, int out_size, void* d_ws, size_t ws_size,
                              hipStream_t stream) {
    (void)in_sizes; (void)n_in; (void)out_size; (void)ws_size;

    const float* x   = (const float*)d_in[0];
    const float* Wu  = (const float*)d_in[1];
    const float* Wc  = (const float*)d_in[2];
    const float* Wh  = (const float*)d_in[3];
    const float* b_f = (const float*)d_in[4];
    const float* v_f = (const float*)d_in[5];
    const float* b_r = (const float*)d_in[6];
    const float* v_r = (const float*)d_in[7];
    const float* b_o = (const float*)d_in[8];
    float* out = (float*)d_out;

    // ---- workspace layout ----
    char* ws = (char*)d_ws;
    size_t off = 0;
    float* xh = (float*)(ws + off);
    off += (size_t)LSEQ * NBATCH * 4 * HSZ * sizeof(float);   // 512 MB
    __bf16* xb  = (__bf16*)(ws + off);
    off += (size_t)NBATCH * LSEQ * ISZ * sizeof(__bf16);      // 32 MB
    __bf16* Wub = (__bf16*)(ws + off);
    off += (size_t)4 * HSZ * ISZ * sizeof(__bf16);            // 4 MB
    __bf16* Wcb = (__bf16*)(ws + off);
    off += (size_t)HSZ * HSZ * sizeof(__bf16);                // 2 MB
    __bf16* Whb = (__bf16*)(ws + off);
    off += (size_t)HSZ * HSZ * sizeof(__bf16);                // 2 MB
    const size_t BH = (size_t)NBATCH * HSZ;
    __bf16* stb = (__bf16*)(ws + off);
    off += 4 * BH * sizeof(__bf16);                           // c/h bf16 double buffers
    unsigned* counter = (unsigned*)(ws + off);
    off += 256;

    __bf16* cb[2] = { stb,          stb + 2 * BH };
    __bf16* hb[2] = { stb + BH,     stb + 3 * BH };

    // ---- init (bf16 state + sync counter; runs every call for graph replay) ----
    bptru_init<<<dim3((unsigned)((BH + 255) / 256)), 256, 0, stream>>>(
        cb[0], hb[0], counter, (int)BH);

    // ---- one-time conversions to bf16 ----
    {
        const int nx = NBATCH * LSEQ * ISZ;
        bptru_f32_to_bf16<<<dim3((nx / 4 + 255) / 256), 256, 0, stream>>>(x, xb, nx);
        const int nu = 4 * HSZ * ISZ;
        bptru_f32_to_bf16<<<dim3((nu / 4 + 255) / 256), 256, 0, stream>>>(Wu, Wub, nu);
        const int nh = HSZ * HSZ;
        bptru_f32_to_bf16<<<dim3((nh / 4 + 255) / 256), 256, 0, stream>>>(Wc, Wcb, nh);
        bptru_f32_to_bf16<<<dim3((nh / 4 + 255) / 256), 256, 0, stream>>>(Wh, Whb, nh);
    }

    // ---- Phase 1: input projection GEMM (WMMA bf16) ----
    bptru_gemm_xu<<<dim3(4 * HSZ / 128, LSEQ * NBATCH / 64), 128, 0, stream>>>(xb, Wub, xh);

    // ---- Phase 2: persistent scan, single launch, LDS-resident weights ----
    const size_t lds_bytes = (size_t)2 * NCOL * LDSW * sizeof(__bf16);  // ~129 KB
    float* tail = out + (size_t)NBATCH * LSEQ * HSZ;
    bptru_scan<<<dim3(SCAN_BLOCKS), 128, lds_bytes, stream>>>(
        xh, cb[0], hb[0], cb[1], hb[1],
        Wcb, Whb, b_f, v_f, b_r, v_r, b_o, out, tail, counter);
}
